// RoPE_MHA_49727131353511
// MI455X (gfx1250) — matrix operand order repaired and checked by reading, not compiled, not run on hardware
//
#include <hip/hip_runtime.h>
#include <hip/hip_bf16.h>
#include <stdint.h>

typedef _Float16 v16h __attribute__((ext_vector_type(16)));
typedef _Float16 v8h  __attribute__((ext_vector_type(8)));
typedef float    v8f  __attribute__((ext_vector_type(8)));

#define NB   2
#define NS   2048
#define NDM  1024
#define NH   16
#define NHS  64
#define N3D  3072

__device__ __forceinline__ v16h cat16(v8h lo, v8h hi) {
  return __builtin_shufflevector(lo, hi, 0,1,2,3,4,5,6,7,8,9,10,11,12,13,14,15);
}
__device__ __forceinline__ v8f wmma16(v16h a, v16h b, v8f c) {
  return __builtin_amdgcn_wmma_f32_16x16x32_f16(false, a, false, b, (short)0, c, false, false);
}
// CDNA5 async global->LDS copy (ASYNCcnt-tracked, bypasses VGPRs)
__device__ __forceinline__ void async_cp_b128(uint32_t lds_addr, const void* gaddr) {
  asm volatile("global_load_async_to_lds_b128 %0, %1, off"
               :: "v"(lds_addr), "v"(gaddr) : "memory");
}
__device__ __forceinline__ uint32_t lds_lo32(const void* p) {
  return (uint32_t)(uintptr_t)p;   // LDS aperture lives in the high 32 bits
}

// ---------------------------------------------------------------- converts
__global__ void cvt_f32_f16(const float* __restrict__ in, _Float16* __restrict__ out, int n) {
  for (int i = blockIdx.x * blockDim.x + threadIdx.x; i < n; i += gridDim.x * blockDim.x)
    out[i] = (_Float16)in[i];
}

// f32 (K x N) -> f16 transposed (N x K), 32x32 LDS tiles (weights, done once)
__global__ __launch_bounds__(256) void cvt_t_f16(const float* __restrict__ in,
                                                 _Float16* __restrict__ out, int K, int N) {
  __shared__ _Float16 t[32][33];
  const int bk = blockIdx.y * 32, bn = blockIdx.x * 32;
  const int tx = threadIdx.x & 31, ty = threadIdx.x >> 5;   // 32 x 8
  #pragma unroll
  for (int i = 0; i < 32; i += 8)
    t[ty + i][tx] = (_Float16)in[(size_t)(bk + ty + i) * N + bn + tx];
  __syncthreads();
  #pragma unroll
  for (int i = 0; i < 32; i += 8)
    out[(size_t)(bn + ty + i) * K + bk + tx] = t[tx][ty + i];
}

// ------------------------------------------------- WMMA GEMM, B pre-transposed
// C(MxN) = A(MxK,f16 row-major) * Bt(NxK,f16 row-major)^T + bias(N)
// Block tile 128x64, 8 waves (4x2), wave tile 32x32. Async double-buffered LDS staging.
__global__ __launch_bounds__(256) void gemm_f16t(
    const _Float16* __restrict__ A, const _Float16* __restrict__ Bt,
    const float* __restrict__ bias, float* __restrict__ C,
    int M, int N, int K)
{
  __shared__ _Float16 As[2][128][40];   // 40-half stride: 16B aligned, conflict-free
  __shared__ _Float16 Bs[2][64][40];
  const int tid  = threadIdx.x;
  const int lane = tid & 31;
  const int wave = tid >> 5;
  const int wm = wave >> 1, wn = wave & 1;
  const int lm = lane & 15, grp = lane >> 4;
  const int mBase = blockIdx.y * 128;
  const int nBase = blockIdx.x * 64;

  // per-thread staging chunk coordinates (16B chunks)
  const int ar0 = tid >> 2,  ac0 = (tid & 3) * 8;          // A chunk 0 (rows 0..63)
  const int ar1 = ar0 + 64;                                 // A chunk 1 (rows 64..127)
  const int br  = tid >> 2,  bc  = (tid & 3) * 8;          // Bt chunk (rows 0..63)
  const _Float16* Ag = A  + (size_t)(mBase + ar0) * K + ac0;
  const _Float16* Ag2= A  + (size_t)(mBase + ar1) * K + ac0;
  const _Float16* Bg = Bt + (size_t)(nBase + br ) * K + bc;

  const int nt = K >> 5;
  // prologue: tile 0 -> buffer 0
  async_cp_b128(lds_lo32(&As[0][ar0][ac0]), Ag);
  async_cp_b128(lds_lo32(&As[0][ar1][ac0]), Ag2);
  async_cp_b128(lds_lo32(&Bs[0][br ][bc ]), Bg);

  v8f acc[2][2] = {};
  const int ka = grp * 8;
  for (int t = 0; t < nt; ++t) {
    const int buf = t & 1;
    if (t + 1 < nt) {  // overlap: stage tile t+1 while computing tile t
      const int k1 = (t + 1) << 5;
      async_cp_b128(lds_lo32(&As[buf ^ 1][ar0][ac0]), Ag  + k1);
      async_cp_b128(lds_lo32(&As[buf ^ 1][ar1][ac0]), Ag2 + k1);
      async_cp_b128(lds_lo32(&Bs[buf ^ 1][br ][bc ]), Bg  + k1);
      asm volatile("s_wait_asynccnt 3" ::: "memory");   // tile t's own copies done
    } else {
      asm volatile("s_wait_asynccnt 0" ::: "memory");
    }
    __syncthreads();                                     // whole tile t resident
    v16h af[2], bf[2];
    #pragma unroll
    for (int mi = 0; mi < 2; ++mi) {
      int m = wm * 32 + mi * 16 + lm;
      af[mi] = cat16(*(const v8h*)&As[buf][m][ka], *(const v8h*)&As[buf][m][16 + ka]);
    }
    #pragma unroll
    for (int ni = 0; ni < 2; ++ni) {
      int n = wn * 32 + ni * 16 + lm;
      bf[ni] = cat16(*(const v8h*)&Bs[buf][n][ka], *(const v8h*)&Bs[buf][n][16 + ka]);
    }
    #pragma unroll
    for (int mi = 0; mi < 2; ++mi)
      #pragma unroll
      for (int ni = 0; ni < 2; ++ni)
        acc[mi][ni] = wmma16(af[mi], bf[ni], acc[mi][ni]);
    __syncthreads();                                     // all reads of tile t done
  }
  // epilogue: one 64-bit base, 32-bit offsets
  float* Cp = C + (size_t)(mBase + wm * 32 + 8 * grp) * N + (nBase + wn * 32 + lm);
  const float* bp = bias + nBase + wn * 32 + lm;
  #pragma unroll
  for (int ni = 0; ni < 2; ++ni) {
    float bv = bp[ni * 16];
    #pragma unroll
    for (int mi = 0; mi < 2; ++mi)
      #pragma unroll
      for (int r = 0; r < 8; ++r)
        Cp[(mi * 16 + r) * N + ni * 16] = acc[mi][ni][r] + bv;
  }
}

// -------------------------------------------- split qkv + RoPE + KV-cache + layouts
__global__ __launch_bounds__(256) void split_rope(
    const float* __restrict__ xproj, const float* __restrict__ cosb,
    const float* __restrict__ sinb, float* __restrict__ kv_out,
    _Float16* __restrict__ qh, _Float16* __restrict__ kh, _Float16* __restrict__ vt)
{
  int idx = blockIdx.x * blockDim.x + threadIdx.x;   // B*S*H*(HS/2) = 2^21 threads
  int p = idx & 31;
  int h = (idx >> 5) & 15;
  int s = (idx >> 9) & 2047;
  int b = idx >> 20;
  size_t row = (size_t)(b * NS + s) * N3D;
  int d = 2 * p;
  float qe = xproj[row + h * NHS + d],            qo = xproj[row + h * NHS + d + 1];
  float ke = xproj[row + NDM + h * NHS + d],      ko = xproj[row + NDM + h * NHS + d + 1];
  float ve = xproj[row + 2 * NDM + h * NHS + d],  vo = xproj[row + 2 * NDM + h * NHS + d + 1];
  float c  = cosb[s * 32 + p], sn = sinb[s * 32 + p];
  float qr = qe * c - qo * sn, qi = qe * sn + qo * c;
  float kr = ke * c - ko * sn, ki = ke * sn + ko * c;
  size_t ko_ = (((size_t)b * NH + h) * NS + s) * NHS + d;       // KV cache (pre-RoPE K)
  kv_out[ko_]     = ke;  kv_out[ko_ + 1] = ko;
  size_t vo_ = ko_ + (size_t)NB * NH * NS * NHS;
  kv_out[vo_]     = ve;  kv_out[vo_ + 1] = vo;
  size_t qo_ = (((size_t)(b * NH + h)) * NS + s) * NHS + d;     // rotated q/k f16
  qh[qo_] = (_Float16)qr;  qh[qo_ + 1] = (_Float16)qi;
  kh[qo_] = (_Float16)kr;  kh[qo_ + 1] = (_Float16)ki;
  size_t vt0 = (((size_t)(b * NH + h)) * NHS + d) * NS + s;     // V transposed f16
  vt[vt0]      = (_Float16)ve;
  vt[vt0 + NS] = (_Float16)vo;
}

// ------------------------------------------------------- flash attention (WMMA)
__global__ __launch_bounds__(128) void attn_kernel(
    const _Float16* __restrict__ qh, const _Float16* __restrict__ kh,
    const _Float16* __restrict__ vt, _Float16* __restrict__ attn_h)
{
  __shared__ _Float16 Pl[4][16][32];
  const int wl   = threadIdx.x >> 5;
  const int lane = threadIdx.x & 31;
  const int lm = lane & 15, grp = lane >> 4;
  const int w  = blockIdx.x * 4 + wl;          // 0 .. B*H*(S/16)-1
  const int qt = w & 127;
  const int bh = w >> 7;
  const int qbase = qt * 16;
  const int ka = grp * 8;

  const _Float16* qrow = qh + ((size_t)bh * NS + qbase + lm) * NHS;
  v16h aq0 = cat16(*(const v8h*)(qrow + ka),      *(const v8h*)(qrow + 16 + ka));
  v16h aq1 = cat16(*(const v8h*)(qrow + 32 + ka), *(const v8h*)(qrow + 48 + ka));

  float rmax[8], rsum[8];
  v8f o[4] = {};
  #pragma unroll
  for (int r = 0; r < 8; ++r) { rmax[r] = -INFINITY; rsum[r] = 0.f; }

  const int kend = qbase + 16;
  for (int kb = 0; kb < kend; kb += 32) {
    const _Float16* krow0 = kh + ((size_t)bh * NS + kb + lm) * NHS;
    const _Float16* krow1 = krow0 + 16 * NHS;
    v8f c0 = {}, c1 = {};
    c0 = wmma16(aq0, cat16(*(const v8h*)(krow0 + ka),      *(const v8h*)(krow0 + 16 + ka)),      c0);
    c0 = wmma16(aq1, cat16(*(const v8h*)(krow0 + 32 + ka), *(const v8h*)(krow0 + 48 + ka)), c0);
    c1 = wmma16(aq0, cat16(*(const v8h*)(krow1 + ka),      *(const v8h*)(krow1 + 16 + ka)),      c1);
    c1 = wmma16(aq1, cat16(*(const v8h*)(krow1 + 32 + ka), *(const v8h*)(krow1 + 48 + ka)), c1);
    #pragma unroll
    for (int r = 0; r < 8; ++r) {            // scale + causal mask (MIN_FP16 semantics)
      int row = qbase + r + 8 * grp;
      float s0 = c0[r] * 0.125f, s1 = c1[r] * 0.125f;
      if (kb + lm > row)      s0 = -65504.f;
      if (kb + 16 + lm > row) s1 = -65504.f;
      c0[r] = s0; c1[r] = s1;
    }
    float e0[8], e1[8];
    #pragma unroll
    for (int r = 0; r < 8; ++r) {            // online softmax (row = 16 lanes of a half)
      float v = fmaxf(c0[r], c1[r]);
      #pragma unroll
      for (int m = 1; m <= 8; m <<= 1) v = fmaxf(v, __shfl_xor(v, m, 32));
      float mnew = fmaxf(v, rmax[r]);
      float scale = __expf(rmax[r] - mnew);
      e0[r] = __expf(c0[r] - mnew);
      e1[r] = __expf(c1[r] - mnew);
      float ss = e0[r] + e1[r];
      #pragma unroll
      for (int m = 1; m <= 8; m <<= 1) ss += __shfl_xor(ss, m, 32);
      rsum[r] = rsum[r] * scale + ss;
      rmax[r] = mnew;
      #pragma unroll
      for (int t = 0; t < 4; ++t) o[t][r] *= scale;
    }
    #pragma unroll
    for (int r = 0; r < 8; ++r) {            // stage P through LDS for the A-fragment
      Pl[wl][r + 8 * grp][lm]      = (_Float16)e0[r];
      Pl[wl][r + 8 * grp][16 + lm] = (_Float16)e1[r];
    }
    asm volatile("s_wait_dscnt 0" ::: "memory");
    v16h ap = cat16(*(const v8h*)&Pl[wl][lm][ka], *(const v8h*)&Pl[wl][lm][16 + ka]);
    #pragma unroll
    for (int t = 0; t < 4; ++t) {            // P @ V
      const _Float16* vrow = vt + ((size_t)bh * NHS + t * 16 + lm) * NS + kb + ka;
      v16h bv = cat16(*(const v8h*)vrow, *(const v8h*)(vrow + 16));
      o[t] = wmma16(ap, bv, o[t]);
    }
  }
  const int b = bh >> 4, h = bh & 15;
  _Float16* op = attn_h + ((size_t)b * NS + qbase + 8 * grp) * NDM + h * NHS + lm;
  #pragma unroll
  for (int t = 0; t < 4; ++t)
    #pragma unroll
    for (int r = 0; r < 8; ++r)
      op[r * NDM + t * 16] = (_Float16)(o[t][r] / rsum[r]);
}

// ---------------------------------------------------------------- launcher
extern "C" void kernel_launch(void* const* d_in, const int* in_sizes, int n_in,
                              void* d_out, int out_size, void* d_ws, size_t ws_size,
                              hipStream_t stream) {
  const float* x      = (const float*)d_in[0];
  const float* cos_q  = (const float*)d_in[1];
  const float* sin_q  = (const float*)d_in[2];
  const float* W_proj = (const float*)d_in[6];
  const float* b_proj = (const float*)d_in[7];
  const float* W_ff   = (const float*)d_in[8];
  const float* b_ff   = (const float*)d_in[9];

  char* ws = (char*)d_ws;
  size_t off = 0;
  _Float16* xh      = (_Float16*)(ws + off); off += (size_t)NB * NS * NDM * 2;
  _Float16* wprojt  = (_Float16*)(ws + off); off += (size_t)NDM * N3D * 2;     // (3072 x 1024)
  _Float16* wfft    = (_Float16*)(ws + off); off += (size_t)NDM * NDM * 2;     // (1024 x 1024)
  float*    xproj   = (float*)   (ws + off); off += (size_t)NB * NS * N3D * 4;
  _Float16* qh      = (_Float16*)(ws + off); off += (size_t)NB * NH * NS * NHS * 2;
  _Float16* kh      = (_Float16*)(ws + off); off += (size_t)NB * NH * NS * NHS * 2;
  _Float16* vt      = (_Float16*)(ws + off); off += (size_t)NB * NH * NHS * NS * 2;
  _Float16* attn_h  = xh;   // x dead after QKV GEMM: reuse

  float* ff_out = (float*)d_out;
  float* kv_out = (float*)d_out + (size_t)NB * NS * NDM;

  // 1) staging: x -> f16 ; weights -> f16 transposed (N x K)
  cvt_f32_f16<<<4096, 256, 0, stream>>>(x, xh, NB * NS * NDM);
  cvt_t_f16<<<dim3(N3D / 32, NDM / 32), 256, 0, stream>>>(W_proj, wprojt, NDM, N3D);
  cvt_t_f16<<<dim3(NDM / 32, NDM / 32), 256, 0, stream>>>(W_ff,  wfft,  NDM, NDM);

  // 2) QKV projection
  gemm_f16t<<<dim3(N3D / 64, (NB * NS) / 128), 256, 0, stream>>>(
      xh, wprojt, b_proj, xproj, NB * NS, N3D, NDM);

  // 3) split + RoPE + KV cache + layouts
  split_rope<<<(NB * NS * NH * (NHS / 2)) / 256, 256, 0, stream>>>(
      xproj, cos_q, sin_q, kv_out, qh, kh, vt);

  // 4) flash attention
  attn_kernel<<<(NB * NH * (NS / 16)) / 4, 128, 0, stream>>>(qh, kh, vt, attn_h);

  // 5) FF projection -> d_out
  gemm_f16t<<<dim3(NDM / 64, (NB * NS) / 128), 256, 0, stream>>>(
      attn_h, wfft, b_ff, ff_out, NB * NS, NDM, NDM);
}
